// MultiRelationalGCNLayer_83330955477834
// MI455X (gfx1250) — matrix operand order repaired and checked by reading, not compiled, not run on hardware
//
#include <hip/hip_runtime.h>
#include <hip/hip_bf16.h>

// ---------------------------------------------------------------------------
// Multi-relational GCN layer for gfx1250 (MI455X).
//   Pass 1: convert X and [W_r0|W_r1|W_r2|self_W] to bf16 once (streaming).
//   Pass 2: H = Xbf @ Wbf^T via v_wmma_f32_16x16x32_bf16 (pure b128 loads).
//   Pass 3: out = relu( sum_r scatter(H_r[src]/deg[dst]) + H_self + b )
// H (numNodes x 512 fp32 ~= 98 MB) lives in d_ws and fits in the 192 MB L2,
// so edge-phase gathers + fp32 atomics stay on-chip.
// ---------------------------------------------------------------------------

typedef __attribute__((ext_vector_type(16))) __bf16 v16bf;
typedef __attribute__((ext_vector_type(8)))  float  v8f;

#define IN_DIM  128
#define H_COLS  512   // 4 * 128 : [r0 | r1 | r2 | self]

static __device__ __forceinline__ unsigned int f2bf(float f) {
    // round-to-nearest-even fp32 -> bf16 (returned in low 16 bits)
    unsigned int x = __float_as_uint(f);
    unsigned int r = x + 0x7FFFu + ((x >> 16) & 1u);
    return r >> 16;
}

union ABPack { v16bf v; uint4 q[2]; };

// ---------------------------------------------------------------------------
// Kernel 0: streaming fp32 -> bf16 (2 elements / thread, packed dword store)
// ---------------------------------------------------------------------------
__global__ void k_f32_to_bf16(const float* __restrict__ in,
                              unsigned int* __restrict__ out, int nPairs) {
    int i = blockIdx.x * blockDim.x + threadIdx.x;
    if (i < nPairs) {
        float2 f = *(const float2*)(in + 2 * (size_t)i);
        out[i] = f2bf(f.x) | (f2bf(f.y) << 16);
    }
}

// ---------------------------------------------------------------------------
// Kernel 1: zero a float buffer (degree counters)
// ---------------------------------------------------------------------------
__global__ void k_zero_f32(float* __restrict__ p, int n) {
    int i = blockIdx.x * blockDim.x + threadIdx.x;
    if (i < n) p[i] = 0.0f;
}

// ---------------------------------------------------------------------------
// Kernel 2: degree count per relation: deg[dst[e]] += 1
// ---------------------------------------------------------------------------
__global__ void k_count_deg(float* __restrict__ deg, const int* __restrict__ dst, int nEdges) {
    int e = blockIdx.x * blockDim.x + threadIdx.x;
    if (e < nEdges) atomicAdd(&deg[dst[e]], 1.0f);
}

// ---------------------------------------------------------------------------
// Kernel 3: WMMA GEMM  H[numNodes x 512] = Xbf[numNodes x 128] @ Wbf^T
//   One wave per 16x16 output tile; K = 128 as 4 steps of bf16 16x16x32.
//   Inputs are pre-converted bf16, so the inner loop is b128 loads + wmma.
//   Lane layouts per CDNA5 ISA 7.12.2:
//     A 16x32 bf16: lanes 0-15 row M=l16 K={0..7,16..23}; lanes 16-31 K={8..15,24..31}
//     B 32x16 bf16: col = l16; lanes 0-15 K={0..7,16..23}; lanes 16-31 K={8..15,24..31}
//     D f32:        VGPR i -> row tile+half*8+i, col = l16
// ---------------------------------------------------------------------------
__global__ void k_gemm_wmma(const unsigned short* __restrict__ Xbf,
                            const unsigned short* __restrict__ Wbf,  // 512 x 128
                            float* __restrict__ H,
                            int numNodes, int tilesM) {
    const int waveInBlk = threadIdx.x >> 5;
    const int lane      = threadIdx.x & 31;
    const int waveId    = blockIdx.x * (blockDim.x >> 5) + waveInBlk;

    const int tileN = waveId & 31;        // 32 tiles of 16 cols -> 512 cols
    const int tileM = waveId >> 5;        // node tiles
    if (tileM >= tilesM) return;

    const int half = lane >> 4;           // 0: lanes 0-15, 1: lanes 16-31
    const int l16  = lane & 15;

    int row = tileM * 16 + l16;
    if (row >= numNodes) row = numNodes - 1;     // clamp (tiles are exact for N=50000)

    const int ncol = tileN * 16 + l16;           // global output column 0..511

    const unsigned short* ap = Xbf + (size_t)row  * IN_DIM;
    const unsigned short* bp = Wbf + (size_t)ncol * IN_DIM;

    v8f acc = {};

    #pragma unroll
    for (int s = 0; s < 4; ++s) {
        const int k0 = s * 32;

        ABPack A;   // a[0..7]=K k0+off.., a[8..15]=K k0+16+off..  (off = half*8)
        A.q[0] = *(const uint4*)(ap + k0 + half * 8);
        A.q[1] = *(const uint4*)(ap + k0 + 16 + half * 8);

        ABPack B;   // b[0..7]=K k0+off.., b[8..15]=K k0+16+off..  (off = half*8)
        B.q[0] = *(const uint4*)(bp + k0 + half * 8);
        B.q[1] = *(const uint4*)(bp + k0 + 16 + half * 8);

        acc = __builtin_amdgcn_wmma_f32_16x16x32_bf16(
            /*neg_a=*/false, A.v, /*neg_b=*/false, B.v,
            /*c_mod=*/(short)0, acc, /*reuse_a=*/false, /*reuse_b=*/false);
    }

    // ---- store D: VGPR i -> row outRowBase+i, col ncol ----
    // Single whole-tile bounds check so the hot path is 8 plain b32 stores
    // with immediate offsets off one base address.
    const int outRowBase = tileM * 16 + half * 8;
    float* hp = H + (size_t)outRowBase * H_COLS + ncol;
    if (outRowBase + 7 < numNodes) {
        #pragma unroll
        for (int i = 0; i < 8; ++i) hp[(size_t)i * H_COLS] = acc[i];
    } else {
        #pragma unroll
        for (int i = 0; i < 8; ++i)
            if (outRowBase + i < numNodes) hp[(size_t)i * H_COLS] = acc[i];
    }
}

// ---------------------------------------------------------------------------
// Kernel 4: out = H_self + bias   (self slice = cols 384..511 of H)
//   float4 per thread: i indexes groups of 4 channels.
// ---------------------------------------------------------------------------
__global__ void k_init_out(const float* __restrict__ H,
                           const float* __restrict__ bias,
                           float* __restrict__ out, int totalVec4) {
    int i = blockIdx.x * blockDim.x + threadIdx.x;
    if (i >= totalVec4) return;
    const int n = i >> 5;                 // node (32 float4 groups per node)
    const int c = (i & 31) * 4;           // channel base
    const float4 h = *(const float4*)(H + (size_t)n * H_COLS + 384 + c);
    const float4 b = *(const float4*)(bias + c);
    float4 o;
    o.x = h.x + b.x; o.y = h.y + b.y; o.z = h.z + b.z; o.w = h.w + b.w;
    *(float4*)(out + (size_t)n * IN_DIM + c) = o;
}

// ---------------------------------------------------------------------------
// Kernel 5: edge scatter.  One wave per edge; each lane moves 4 floats.
//   out[dst] += H[src, colOff:colOff+128] / max(deg[dst], 1)
// ---------------------------------------------------------------------------
__global__ void k_scatter(const float* __restrict__ H, int colOff,
                          const float* __restrict__ deg,
                          const int* __restrict__ src,
                          const int* __restrict__ dst,
                          float* __restrict__ out, int nEdges) {
    const int wid  = (blockIdx.x * blockDim.x + threadIdx.x) >> 5;
    const int lane = threadIdx.x & 31;
    if (wid >= nEdges) return;

    const int s = src[wid];
    const int d = dst[wid];
    float dg = deg[d];
    if (dg < 1.0f) dg = 1.0f;
    const float inv = 1.0f / dg;

    const float4 v = *(const float4*)(H + (size_t)s * H_COLS + colOff + lane * 4);
    float* op = out + (size_t)d * IN_DIM + lane * 4;
    atomicAdd(op + 0, v.x * inv);
    atomicAdd(op + 1, v.y * inv);
    atomicAdd(op + 2, v.z * inv);
    atomicAdd(op + 3, v.w * inv);
}

// ---------------------------------------------------------------------------
// Kernel 6: in-place ReLU on out (float4)
// ---------------------------------------------------------------------------
__global__ void k_relu(float* __restrict__ out, int totalVec4) {
    int i = blockIdx.x * blockDim.x + threadIdx.x;
    if (i >= totalVec4) return;
    float4 v = *(float4*)(out + 4 * (size_t)i);
    v.x = v.x > 0.0f ? v.x : 0.0f;
    v.y = v.y > 0.0f ? v.y : 0.0f;
    v.z = v.z > 0.0f ? v.z : 0.0f;
    v.w = v.w > 0.0f ? v.w : 0.0f;
    *(float4*)(out + 4 * (size_t)i) = v;
}

// ---------------------------------------------------------------------------
extern "C" void kernel_launch(void* const* d_in, const int* in_sizes, int n_in,
                              void* d_out, int out_size, void* d_ws, size_t ws_size,
                              hipStream_t stream) {
    const float* X     = (const float*)d_in[0];
    const float* W0    = (const float*)d_in[1];
    const float* W1    = (const float*)d_in[2];
    const float* W2    = (const float*)d_in[3];
    const float* Wself = (const float*)d_in[4];
    const float* bias  = (const float*)d_in[5];
    const int* src0 = (const int*)d_in[6];
    const int* dst0 = (const int*)d_in[7];
    const int* src1 = (const int*)d_in[8];
    const int* dst1 = (const int*)d_in[9];
    const int* src2 = (const int*)d_in[10];
    const int* dst2 = (const int*)d_in[11];

    const int numNodes = in_sizes[0] / IN_DIM;
    const int E0 = in_sizes[6];
    const int E1 = in_sizes[8];
    const int E2 = in_sizes[10];

    // workspace layout (all 16B-aligned):
    //   H   : numNodes*512 f32
    //   deg : 3*numNodes   f32
    //   Xbf : numNodes*128 bf16
    //   Wbf : 512*128      bf16   ([W0|W1|W2|Wself] stacked row-major)
    float* H   = (float*)d_ws;
    float* deg = H + (size_t)numNodes * H_COLS;
    float* deg0 = deg;
    float* deg1 = deg + numNodes;
    float* deg2 = deg + 2 * (size_t)numNodes;
    unsigned short* Xbf = (unsigned short*)(deg + 3 * (size_t)numNodes);
    unsigned short* Wbf = Xbf + (size_t)numNodes * IN_DIM;

    const int THREADS = 256;

    // 0) one-shot bf16 conversion of X and the four weight matrices
    {
        int nPairs = numNodes * IN_DIM / 2;
        k_f32_to_bf16<<<(nPairs + THREADS - 1) / THREADS, THREADS, 0, stream>>>(
            X, (unsigned int*)Xbf, nPairs);
        int wPairs = IN_DIM * IN_DIM / 2;   // 8192 dwords per matrix
        k_f32_to_bf16<<<(wPairs + THREADS - 1) / THREADS, THREADS, 0, stream>>>(
            W0, (unsigned int*)(Wbf + 0 * 128 * 128), wPairs);
        k_f32_to_bf16<<<(wPairs + THREADS - 1) / THREADS, THREADS, 0, stream>>>(
            W1, (unsigned int*)(Wbf + 1 * 128 * 128), wPairs);
        k_f32_to_bf16<<<(wPairs + THREADS - 1) / THREADS, THREADS, 0, stream>>>(
            W2, (unsigned int*)(Wbf + 2 * 128 * 128), wPairs);
        k_f32_to_bf16<<<(wPairs + THREADS - 1) / THREADS, THREADS, 0, stream>>>(
            Wself, (unsigned int*)(Wbf + 3 * 128 * 128), wPairs);
    }

    // 1) zero degree counters
    {
        int n = 3 * numNodes;
        k_zero_f32<<<(n + THREADS - 1) / THREADS, THREADS, 0, stream>>>(deg, n);
    }

    // 2) degree counts
    k_count_deg<<<(E0 + THREADS - 1) / THREADS, THREADS, 0, stream>>>(deg0, dst0, E0);
    k_count_deg<<<(E1 + THREADS - 1) / THREADS, THREADS, 0, stream>>>(deg1, dst1, E1);
    k_count_deg<<<(E2 + THREADS - 1) / THREADS, THREADS, 0, stream>>>(deg2, dst2, E2);

    // 3) WMMA GEMM: H = Xbf @ Wbf^T
    {
        const int tilesM      = (numNodes + 15) / 16;
        const int totalWaves  = tilesM * 32;          // 32 col-tiles of 16
        const int wavesPerBlk = THREADS / 32;         // 8 waves / block
        const int blocks = (totalWaves + wavesPerBlk - 1) / wavesPerBlk;
        k_gemm_wmma<<<blocks, THREADS, 0, stream>>>(Xbf, Wbf, H, numNodes, tilesM);
    }

    // 4) out = H_self + bias
    {
        int totalVec4 = numNodes * IN_DIM / 4;
        k_init_out<<<(totalVec4 + THREADS - 1) / THREADS, THREADS, 0, stream>>>(
            H, bias, (float*)d_out, totalVec4);
    }

    // 5) scatter the three relations (one wave per edge)
    {
        const int edgesPerBlk = THREADS / 32;
        k_scatter<<<(E0 + edgesPerBlk - 1) / edgesPerBlk, THREADS, 0, stream>>>(
            H, 0,   deg0, src0, dst0, (float*)d_out, E0);
        k_scatter<<<(E1 + edgesPerBlk - 1) / edgesPerBlk, THREADS, 0, stream>>>(
            H, 128, deg1, src1, dst1, (float*)d_out, E1);
        k_scatter<<<(E2 + edgesPerBlk - 1) / edgesPerBlk, THREADS, 0, stream>>>(
            H, 256, deg2, src2, dst2, (float*)d_out, E2);
    }

    // 6) ReLU
    {
        int totalVec4 = numNodes * IN_DIM / 4;
        k_relu<<<(totalVec4 + THREADS - 1) / THREADS, THREADS, 0, stream>>>(
            (float*)d_out, totalVec4);
    }
}
